// Attention_81973745811710
// MI455X (gfx1250) — matrix operand order repaired and checked by reading, not compiled, not run on hardware
//
#include <hip/hip_runtime.h>

typedef __bf16 bf16_t;
typedef bf16_t bf16x16 __attribute__((ext_vector_type(16)));
typedef bf16_t bf16x8  __attribute__((ext_vector_type(8)));
typedef bf16_t bf16x4  __attribute__((ext_vector_type(4)));
typedef float  f32x8   __attribute__((ext_vector_type(8)));
typedef unsigned int u32x4 __attribute__((ext_vector_type(4)));
typedef int i32x4 __attribute__((ext_vector_type(4)));
typedef int i32x8 __attribute__((ext_vector_type(8)));

#define NHEADS  16
#define DHEAD   64
#define DMODEL  1024
#define BATCH   4
#define SEQ     2048

#define GLDS 40   // padded LDS stride for 32-wide bf16 tiles: 80B = 20 banks
#define KTS  72   // padded stride for 64-wide bf16 tile rows: 144B
#define VTS  40
#define PTS  40

// TDM path (this toolchain: clang-23 / therock headers -> 6-arg builtin).
#if defined(__HIP_DEVICE_COMPILE__) && \
    __has_builtin(__builtin_amdgcn_tensor_load_to_lds) && \
    __has_builtin(__builtin_amdgcn_s_wait_tensorcnt)
#define USE_TDM 1
#else
#define USE_TDM 0
#endif

#define WMMA_BF16(a,b,c) \
  __builtin_amdgcn_wmma_f32_16x16x32_bf16(false,(a),false,(b),(short)0,(c),false,false)

__device__ __forceinline__ bf16x16 frag_cat(bf16x8 lo, bf16x8 hi) {
  return __builtin_shufflevector(lo, hi, 0,1,2,3,4,5,6,7,8,9,10,11,12,13,14,15);
}
// A-fragment (16x32, 16-bit): lane = row M=lane%16; g=lane/16 selects
// K chunks {kbase+g*8..+7} and {kbase+16+g*8..+7}   (ISA 7.12.2)
__device__ __forceinline__ bf16x16 ldA(const bf16_t* row, int g, int kbase) {
  bf16x8 lo = *(const bf16x8*)(row + kbase + g * 8);
  bf16x8 hi = *(const bf16x8*)(row + kbase + 16 + g * 8);
  return frag_cat(lo, hi);
}
// B-fragment (32x16, 16-bit): lane = col N=lane%16; g=lane/16 selects
// K chunks {kbase+g*8..+7} and {kbase+16+g*8..+7}
__device__ __forceinline__ bf16x16 ldB(const bf16_t* row, int g, int kbase) {
  bf16x8 lo = *(const bf16x8*)(row + kbase + g * 8);
  bf16x8 hi = *(const bf16x8*)(row + kbase + 16 + g * 8);
  return frag_cat(lo, hi);
}
__device__ __forceinline__ f32x8 zero8() {
  f32x8 z = {0.f,0.f,0.f,0.f,0.f,0.f,0.f,0.f};
  return z;
}

#if USE_TDM
// 2D tile load via Tensor Data Mover. D# packing per ISA ch.8 (CDNA5):
// group0: count=1 | lds_addr | global_addr | type=2
// group1: data_size=8B, pad ctrl | tensor dims == tile dims | dim0 stride
// Tile width d0 and stride in 8-byte units; pad_amt==0 disables padding.
__device__ __forceinline__ void tdm_load_2d(unsigned lds_off, const void* gptr,
                                            unsigned d0_units, unsigned d1_rows,
                                            unsigned stride_units,
                                            unsigned pad_iv, unsigned pad_amt) {
  unsigned long long ga = (unsigned long long)(size_t)gptr;
  u32x4 g0;
  g0[0] = 1u;                                        // count=1, user descriptor
  g0[1] = lds_off;                                   // lds_addr (bytes)
  g0[2] = (unsigned)ga;                              // global_addr[31:0]
  g0[3] = (unsigned)((ga >> 32) & 0x1FFFFFFull) | (2u << 30); // ga[56:32], type=2
  unsigned w0 = 3u << 16;                            // data_size = 8B
  if (pad_amt) w0 |= (1u << 20) | (pad_iv << 22) | (pad_amt << 25);
  i32x8 g1;
  g1[0] = (int)w0;
  g1[1] = (int)(d0_units << 16);                     // tensor_dim0[15:0]
  g1[2] = (int)(d1_rows << 16);                      // tensor_dim0 hi=0 | tensor_dim1 lo
  g1[3] = (int)(d0_units << 16);                     // tensor_dim1 hi=0 | tile_dim0
  g1[4] = (int)(d1_rows & 0xFFFFu);                  // tile_dim1 (tile_dim2=0)
  g1[5] = (int)stride_units;                         // tensor_dim0_stride[31:0]
  g1[6] = 0;                                         // stride hi / dim1_stride lo
  g1[7] = 0;
  i32x4 gz4 = {0, 0, 0, 0};
  i32x8 gz8 = {0, 0, 0, 0, 0, 0, 0, 0};
  // 6-arg form on this toolchain: (g0, g1, g2, g3, <extra group, zero>, cpol)
  __builtin_amdgcn_tensor_load_to_lds(g0, g1, gz4, gz4, gz8, 0);
}
__device__ __forceinline__ unsigned lds_off(const void* p) {
  return (unsigned)(size_t)p;                        // flat LDS addr truncates to offset
}
#endif

// ---------------------------------------------------------------- fp32 -> bf16
__global__ void cvt_f32_bf16(const float* __restrict__ src,
                             bf16_t* __restrict__ dst, int n) {
  int i = (blockIdx.x * 256 + threadIdx.x) * 4;
  if (i + 3 < n) {
    float4 v = *(const float4*)(src + i);
    bf16x4 o = {(bf16_t)v.x, (bf16_t)v.y, (bf16_t)v.z, (bf16_t)v.w};
    *(bf16x4*)(dst + i) = o;
  }
}

// ------------------------------------------------- shared 64x64 GEMM core
// C[64x64] = A[64xK] * (B^T[64xK])^T, K=1024, bf16 in / f32 acc.
// 8 waves: wave (wr,wc) computes rows wr*16..+15, cols wc*32..+31 (two WMMAs).
// lA/lB hold two 64xGLDS buffers (double-buffered TDM pipeline).
__device__ __forceinline__ void gemm64x64(const bf16_t* __restrict__ Arows,
                                          const bf16_t* __restrict__ Brows,
                                          bf16_t* lA, bf16_t* lB,
                                          f32x8 acc[2]) {
  const int tid  = threadIdx.x;
  const int lane = tid & 31, wave = tid >> 5;
  const int g = lane >> 4, lm = lane & 15;
  const int wr = wave >> 1, wc = wave & 1;
  acc[0] = zero8(); acc[1] = zero8();
#if USE_TDM
  // tiles: 64 rows x 32 bf16 (= 8 units of 8B), row stride 256 units,
  // pad 4 DW after every 16 DW -> LDS stride GLDS=40 elems.
  if (wave == 0) tdm_load_2d(lds_off(lA), Arows, 8, 64, 256, 3, 3);
  if (wave == 1) tdm_load_2d(lds_off(lB), Brows, 8, 64, 256, 3, 3);
  int buf = 0;
  for (int k0 = 0; k0 < DMODEL; k0 += 32) {
    __builtin_amdgcn_s_wait_tensorcnt(0);   // tile k0 landed (issuing waves)
    __syncthreads();                        // publish to all waves
    if (k0 + 32 < DMODEL) {                 // prefetch next tile into other buffer
      int nb = (buf ^ 1) * 64 * GLDS;
      if (wave == 0) tdm_load_2d(lds_off(lA + nb), Arows + k0 + 32, 8, 64, 256, 3, 3);
      if (wave == 1) tdm_load_2d(lds_off(lB + nb), Brows + k0 + 32, 8, 64, 256, 3, 3);
    }
    const bf16_t* cA = lA + buf * 64 * GLDS;
    const bf16_t* cB = lB + buf * 64 * GLDS;
    bf16x16 af = ldA(cA + (wr * 16 + lm) * GLDS, g, 0);
    bf16x16 b0 = ldB(cB + (wc * 32 + lm) * GLDS, g, 0);
    bf16x16 b1 = ldB(cB + (wc * 32 + 16 + lm) * GLDS, g, 0);
    acc[0] = WMMA_BF16(af, b0, acc[0]);
    acc[1] = WMMA_BF16(af, b1, acc[1]);
    buf ^= 1;
  }
#else
  const int lrow = tid >> 2, lch = (tid & 3) * 8;
  for (int k0 = 0; k0 < DMODEL; k0 += 32) {
    __syncthreads();
    *(bf16x8*)(lA + lrow * GLDS + lch) =
        *(const bf16x8*)(Arows + lrow * DMODEL + k0 + lch);
    *(bf16x8*)(lB + lrow * GLDS + lch) =
        *(const bf16x8*)(Brows + lrow * DMODEL + k0 + lch);
    __syncthreads();
    if (k0 + 32 < DMODEL) {
      __builtin_prefetch(Arows + lrow * DMODEL + k0 + 32 + lch, 0, 3);
      __builtin_prefetch(Brows + lrow * DMODEL + k0 + 32 + lch, 0, 3);
    }
    bf16x16 af = ldA(lA + (wr * 16 + lm) * GLDS, g, 0);
    bf16x16 b0 = ldB(lB + (wc * 32 + lm) * GLDS, g, 0);
    bf16x16 b1 = ldB(lB + (wc * 32 + 16 + lm) * GLDS, g, 0);
    acc[0] = WMMA_BF16(af, b0, acc[0]);
    acc[1] = WMMA_BF16(af, b1, acc[1]);
  }
#endif
}

// ------------------------------------------------- QKV projection
// Y[r, c] = x[r,:] . W[c,:]  (r = b*SEQ+t, c in 0..3071; W = [Q;K;V] rows)
// scatter: q,k -> [b,h,S,64] bf16; v -> transposed [b,h,64,S] bf16
__global__ void qkv_gemm(const bf16_t* __restrict__ xh,
                         const bf16_t* __restrict__ W,
                         bf16_t* __restrict__ qbuf,
                         bf16_t* __restrict__ kbuf,
                         bf16_t* __restrict__ vtb) {
  __shared__ __align__(16) bf16_t lA[2 * 64 * GLDS];
  __shared__ __align__(16) bf16_t lB[2 * 64 * GLDS];
  f32x8 acc[2];
  gemm64x64(xh + (size_t)blockIdx.x * 64 * DMODEL,
            W  + (size_t)blockIdx.y * 64 * DMODEL, lA, lB, acc);
  const int tid = threadIdx.x, lane = tid & 31, wave = tid >> 5;
  const int g = lane >> 4, lm = lane & 15;
  const int wr = wave >> 1, wc = wave & 1;
  const int rbase = blockIdx.x * 64 + wr * 16 + 8 * g;
  const int cbase = blockIdx.y * 64 + wc * 32 + lm;
#pragma unroll
  for (int t = 0; t < 2; ++t) {
#pragma unroll
    for (int j = 0; j < 8; ++j) {
      int r = rbase + j;
      int c = cbase + t * 16;
      int b = r >> 11, tok = r & (SEQ - 1);
      int which = c >> 10, cc = c & 1023;
      int n = cc >> 6, h = cc & 63;
      int bh = b * NHEADS + n;
      bf16_t bv = (bf16_t)acc[t][j];
      if (which == 0)      qbuf[((size_t)bh * SEQ + tok) * DHEAD + h] = bv;
      else if (which == 1) kbuf[((size_t)bh * SEQ + tok) * DHEAD + h] = bv;
      else                 vtb [((size_t)bh * DHEAD + h) * SEQ + tok] = bv;
    }
  }
}

// ------------------------------------------------- flash attention (causal)
// One block = 128 queries of one (b,h); each wave owns 16 queries.
__global__ void flash_attn(const bf16_t* __restrict__ qbuf,
                           const bf16_t* __restrict__ kbuf,
                           const bf16_t* __restrict__ vtb,
                           bf16_t* __restrict__ zb) {
  __shared__ __align__(16) bf16_t Kt[2][32 * KTS];     // 32 keys x 64 dims
  __shared__ __align__(16) bf16_t Vt[2][64 * VTS];     // 64 dims x 32 keys
  __shared__ __align__(16) bf16_t Pt[8 * 16 * PTS];    // per-wave P staging
  const int bh = blockIdx.y;
  const int b = bh >> 4, h = bh & 15;
  const int tid = threadIdx.x, lane = tid & 31, wave = tid >> 5;
  const int g = lane >> 4, lm = lane & 15;
  const int qb0 = blockIdx.x * 128 + wave * 16;

  const bf16_t* qr = qbuf + ((size_t)bh * SEQ + qb0 + lm) * DHEAD;
  bf16x16 aq0 = ldA(qr, g, 0);
  bf16x16 aq1 = ldA(qr, g, 32);

  f32x8 accZ[4];
  float mrow[8], lrow[8];
#pragma unroll
  for (int d = 0; d < 4; ++d) accZ[d] = zero8();
#pragma unroll
  for (int j = 0; j < 8; ++j) { mrow[j] = -3.0e38f; lrow[j] = 0.f; }

  bf16_t* Pw = Pt + wave * 16 * PTS;
  const bf16_t* Kbase = kbuf + (size_t)bh * SEQ * DHEAD;
  const bf16_t* Vbase = vtb  + (size_t)bh * DHEAD * SEQ;
  const int kend = blockIdx.x * 128 + 128;
#if !USE_TDM
  const int ldKrow = tid >> 3, ldKch = (tid & 7) * 8;
  const int ldVrow = tid >> 2, ldVch = (tid & 3) * 8;
#endif

#if USE_TDM
  // K tile: 32 rows x 128B (16 units), contiguous rows -> stride 16; pad 4DW/32DW (KTS=72)
  // V tile: 64 rows x 64B  (8 units), row stride SEQ*2B = 512 units; pad 4DW/16DW (VTS=40)
  if (wave == 0) tdm_load_2d(lds_off(&Kt[0][0]), Kbase, 16, 32, 16, 4, 3);
  if (wave == 1) tdm_load_2d(lds_off(&Vt[0][0]), Vbase, 8, 64, 512, 3, 3);
  int bufc = 0;
#endif
  for (int kc = 0; kc < kend; kc += 32) {
#if USE_TDM
    __builtin_amdgcn_s_wait_tensorcnt(0);
    __syncthreads();
    if (kc + 32 < kend) {
      if (wave == 0)
        tdm_load_2d(lds_off(&Kt[bufc ^ 1][0]),
                    Kbase + (size_t)(kc + 32) * DHEAD, 16, 32, 16, 4, 3);
      if (wave == 1)
        tdm_load_2d(lds_off(&Vt[bufc ^ 1][0]),
                    Vbase + kc + 32, 8, 64, 512, 3, 3);
    }
    const bf16_t* Ktc = &Kt[bufc][0];
    const bf16_t* Vtc = &Vt[bufc][0];
    bufc ^= 1;
#else
    __syncthreads();
    *(bf16x8*)(&Kt[0][0] + ldKrow * KTS + ldKch) =
        *(const bf16x8*)(Kbase + (size_t)(kc + ldKrow) * DHEAD + ldKch);
    *(bf16x8*)(&Vt[0][0] + ldVrow * VTS + ldVch) =
        *(const bf16x8*)(Vbase + (size_t)ldVrow * SEQ + kc + ldVch);
    __syncthreads();
    const bf16_t* Ktc = &Kt[0][0];
    const bf16_t* Vtc = &Vt[0][0];
#endif
    if (kc <= qb0 + 15) {          // within this wave's causal horizon
      // scores: 16 q x 32 k, contraction over 64 dims
      f32x8 s0 = zero8(), s1 = zero8();
      {
        const bf16_t* kr0 = Ktc + lm * KTS;
        const bf16_t* kr1 = Ktc + (16 + lm) * KTS;
        s0 = WMMA_BF16(aq0, ldB(kr0, g, 0),  s0);
        s0 = WMMA_BF16(aq1, ldB(kr0, g, 32), s0);
        s1 = WMMA_BF16(aq0, ldB(kr1, g, 0),  s1);
        s1 = WMMA_BF16(aq1, ldB(kr1, g, 32), s1);
      }
      if (kc + 31 > qb0) {         // causal mask on diagonal chunks
#pragma unroll
        for (int j = 0; j < 8; ++j) {
          int q = qb0 + j + 8 * g;
          if (kc + lm > q)      s0[j] = -1e10f;
          if (kc + 16 + lm > q) s1[j] = -1e10f;
        }
      }
      // online softmax; rows live across the 16 lanes of each half-wave
#pragma unroll
      for (int j = 0; j < 8; ++j) {
        float cm = fmaxf(s0[j], s1[j]);
#pragma unroll
        for (int off = 1; off < 16; off <<= 1)
          cm = fmaxf(cm, __shfl_xor(cm, off, 32));
        float mnew  = fmaxf(mrow[j], cm);
        float alpha = __expf(mrow[j] - mnew);
        float p0 = __expf(s0[j] - mnew);
        float p1 = __expf(s1[j] - mnew);
        float ps = p0 + p1;
#pragma unroll
        for (int off = 1; off < 16; off <<= 1)
          ps += __shfl_xor(ps, off, 32);
        lrow[j] = lrow[j] * alpha + ps;
        mrow[j] = mnew;
#pragma unroll
        for (int d = 0; d < 4; ++d) accZ[d][j] *= alpha;
        int prow = j + 8 * g;      // C-layout row -> P staging row
        Pw[prow * PTS + lm]      = (bf16_t)p0;
        Pw[prow * PTS + 16 + lm] = (bf16_t)p1;
      }
      // z += P(16x32) x V(32x64); wave-private LDS, ds ops in-order per wave
      bf16x16 ap = ldA(Pw + lm * PTS, g, 0);
#pragma unroll
      for (int d = 0; d < 4; ++d) {
        bf16x16 bv = ldB(Vtc + (d * 16 + lm) * VTS, g, 0);
        accZ[d] = WMMA_BF16(ap, bv, accZ[d]);
      }
    }
  }
  // normalize and emit z[b*S+q, h*64+d] (bf16)
#pragma unroll
  for (int j = 0; j < 8; ++j) {
    float inv = 1.0f / lrow[j];
    int rtok = b * SEQ + qb0 + j + 8 * g;
#pragma unroll
    for (int d = 0; d < 4; ++d)
      zb[(size_t)rtok * DMODEL + h * DHEAD + d * 16 + lm] =
          (bf16_t)(accZ[d][j] * inv);
  }
}

// ------------------------------------------------- output projection
// out[r, m] = z[r,:] . O[m,:]  (fp32 out)
__global__ void out_gemm(const bf16_t* __restrict__ zb,
                         const bf16_t* __restrict__ Obf,
                         float* __restrict__ out) {
  __shared__ __align__(16) bf16_t lA[2 * 64 * GLDS];
  __shared__ __align__(16) bf16_t lB[2 * 64 * GLDS];
  f32x8 acc[2];
  gemm64x64(zb  + (size_t)blockIdx.x * 64 * DMODEL,
            Obf + (size_t)blockIdx.y * 64 * DMODEL, lA, lB, acc);
  const int tid = threadIdx.x, lane = tid & 31, wave = tid >> 5;
  const int g = lane >> 4, lm = lane & 15;
  const int wr = wave >> 1, wc = wave & 1;
  const int rbase = blockIdx.x * 64 + wr * 16 + 8 * g;
  const int cbase = blockIdx.y * 64 + wc * 32 + lm;
#pragma unroll
  for (int t = 0; t < 2; ++t)
#pragma unroll
    for (int j = 0; j < 8; ++j)
      out[(size_t)(rbase + j) * DMODEL + cbase + t * 16] = acc[t][j];
}

// -------------------------------------------------------------- launcher
extern "C" void kernel_launch(void* const* d_in, const int* in_sizes, int n_in,
                              void* d_out, int out_size, void* d_ws, size_t ws_size,
                              hipStream_t stream) {
  const float* x = (const float*)d_in[0];
  const float* Q = (const float*)d_in[1];
  const float* K = (const float*)d_in[2];
  const float* V = (const float*)d_in[3];
  const float* O = (const float*)d_in[4];
  float* out = (float*)d_out;
  char* ws = (char*)d_ws;

  const size_t MB = 1u << 20;
  bf16_t* xh  = (bf16_t*)(ws);             // 16 MiB (reused as z later)
  bf16_t* W   = (bf16_t*)(ws + 16 * MB);   // 6 MiB  [Q;K;V] packed rows
  bf16_t* Obf = (bf16_t*)(ws + 22 * MB);   // 2 MiB
  bf16_t* qb  = (bf16_t*)(ws + 24 * MB);   // 16 MiB
  bf16_t* kb  = (bf16_t*)(ws + 40 * MB);   // 16 MiB
  bf16_t* vtb = (bf16_t*)(ws + 56 * MB);   // 16 MiB  (total 72 MiB)
  bf16_t* zb  = xh;                        // overlay: xh dead after qkv_gemm

  const int NX = BATCH * SEQ * DMODEL;     // 8388608
  const int NW = NHEADS * DHEAD * DMODEL;  // 1048576
  cvt_f32_bf16<<<NX / 1024, 256, 0, stream>>>(x, xh, NX);
  cvt_f32_bf16<<<NW / 1024, 256, 0, stream>>>(Q, W,          NW);
  cvt_f32_bf16<<<NW / 1024, 256, 0, stream>>>(K, W + NW,     NW);
  cvt_f32_bf16<<<NW / 1024, 256, 0, stream>>>(V, W + 2 * NW, NW);
  cvt_f32_bf16<<<NW / 1024, 256, 0, stream>>>(O, Obf,        NW);

  qkv_gemm <<<dim3(BATCH * SEQ / 64, 3 * DMODEL / 64), 256, 0, stream>>>(xh, W, qb, kb, vtb);
  flash_attn<<<dim3(SEQ / 128, BATCH * NHEADS),        256, 0, stream>>>(qb, kb, vtb, zb);
  out_gemm <<<dim3(BATCH * SEQ / 64, DMODEL / 64),     256, 0, stream>>>(zb, Obf, out);
}
